// NodeUpdateBlock_17394617549525
// MI455X (gfx1250) — matrix operand order repaired and checked by reading, not compiled, not run on hardware
//
#include <hip/hip_runtime.h>
#include <hip/hip_bf16.h>

// ---------------------------------------------------------------------------
// Problem constants (match reference)
// ---------------------------------------------------------------------------
#define MS   64
#define MV   32
#define DN   160      // MS + 3*MV
#define FCIN 128
#define NGRP 16
#define EPSL 1e-5f

typedef _Float16 half_t;
typedef __attribute__((ext_vector_type(16))) _Float16 v16h;
typedef __attribute__((ext_vector_type(8)))  float    v8f;

// f16 weight pool offsets (in half elements), transposed to [Nout][K]
#define O_WSS0 0        // 64 x 192
#define O_WVV0 12288    // 64 x 96
#define O_WSSG 18432    // 32 x 192
#define O_WVVG 24576    // 32 x 96
#define O_WSV1 27648    // 32 x 192
#define O_WVS1 33792    // 32 x 96
#define O_W1   36864    // 64 x 128
#define O_W2   45056    // 64 x 64
#define O_W3   49152    // 96 x 64
#define W_TOTAL 55296

#define WMMA_F16(a, b, c) \
  __builtin_amdgcn_wmma_f32_16x16x32_f16(false, (a), false, (b), (short)0, (c), false, false)

// ---------------------------------------------------------------------------
// Fragment loaders (CDNA5 wave32 WMMA 16x16x32 f16 layouts)
// ---------------------------------------------------------------------------
// A (16xK tile): lane = row (lane&15); k = k0 + (j/4)*16 + (lane>>4)*8 + (j%4)*2 + e
__device__ __forceinline__ v16h frag_A(const _Float16* buf, int stride, int k0, int lane) {
  int r = lane & 15, h = (lane >> 4) & 1;
  union { v16h v; unsigned u[8]; } f;
#pragma unroll
  for (int j = 0; j < 8; ++j) {
    int k = k0 + ((j >> 2) << 4) + (h << 3) + ((j & 3) << 1);
    f.u[j] = *reinterpret_cast<const unsigned*>(buf + r * stride + k);
  }
  return f.v;
}

// B (Kx16 tile) from transposed weights WT[Nout][K]: lane = column;
// k = k0 + (j/4)*16 + (lane>>4)*8 + (j%4)*2 + e
__device__ __forceinline__ v16h frag_B(const _Float16* WT, int K, int n0, int k0, int lane) {
  int n = lane & 15, h = (lane >> 4) & 1;
  const _Float16* row = WT + (size_t)(n0 + n) * K + k0 + (h << 3);
  union { v16h v; unsigned u[8]; } f;
#pragma unroll
  for (int j = 0; j < 8; ++j)
    f.u[j] = *reinterpret_cast<const unsigned*>(row + ((j >> 2) << 4) + ((j & 3) << 1));
  return f.v;
}

// ---------------------------------------------------------------------------
// Weight transpose+convert:  src f32 [K][N]  ->  dst f16 [N][K]
// ---------------------------------------------------------------------------
__global__ void w2h_transpose(const float* __restrict__ src, _Float16* __restrict__ dst,
                              int K, int Nout) {
  int i = blockIdx.x * blockDim.x + threadIdx.x;
  if (i < K * Nout) {
    int n = i / K, k = i - n * K;
    dst[i] = (_Float16)src[k * Nout + n];
  }
}

// ---------------------------------------------------------------------------
// Node pre-stage: s_lin, v_lin, sc_s, sc_v   (block = 96 threads, one node)
// ---------------------------------------------------------------------------
__global__ __launch_bounds__(96) void node_pre(
    const float* __restrict__ node_fea, const float* __restrict__ one_hot,
    const float* __restrict__ lp_Ws, const float* __restrict__ lp_bs,
    const float* __restrict__ lp_Wv,
    const float* __restrict__ sc_Ws, const float* __restrict__ sc_Wv,
    float* __restrict__ s_lin, float* __restrict__ v_lin,
    float* __restrict__ sc_s, float* __restrict__ sc_v) {
  int n = blockIdx.x, t = threadIdx.x;
  __shared__ float s_o[MS];
  __shared__ float v_o[3 * MV];
  __shared__ int spec;
  for (int i = t; i < DN; i += 96) {
    float x = node_fea[(size_t)n * DN + i];
    if (i < MS) s_o[i] = x; else v_o[i - MS] = x;
  }
  if (t == 0) {
    int sp = 0;
#pragma unroll
    for (int s = 0; s < 4; ++s) if (one_hot[(size_t)n * 4 + s] > 0.5f) sp = s;
    spec = sp;
  }
  __syncthreads();
  int sp = spec;
  if (t < MS) {
    float a = 0.f, b = 0.f;
    for (int u = 0; u < MS; ++u) {
      float x = s_o[u];
      a += x * lp_Ws[u * MS + t];
      b += x * sc_Ws[(u * 4 + sp) * MS + t];
    }
    s_lin[(size_t)n * MS + t] = a * 0.125f + lp_bs[t];
    sc_s[(size_t)n * MS + t] = b * 0.0625f;                 // 1/sqrt(64*4)
  }
  {
    int w = t / 3, i = t - 3 * w;
    float a = 0.f, b = 0.f;
    for (int u = 0; u < MV; ++u) {
      float x = v_o[u * 3 + i];
      a += x * lp_Wv[u * MV + w];
      b += x * sc_Wv[(u * 4 + sp) * MV + w];
    }
    v_lin[(size_t)n * 96 + t] = a * 0.17677669529663689f;   // 1/sqrt(32)
    sc_v[(size_t)n * 96 + t] = b * 0.08838834764831845f;    // 1/sqrt(128)
  }
}

// ---------------------------------------------------------------------------
// Edge kernel: one wave (32 threads) handles 16 edges; all GEMMs via WMMA f16
// ---------------------------------------------------------------------------
__global__ __launch_bounds__(32) void edge_kernel(
    const float* __restrict__ s_lin, const float* __restrict__ v_lin,
    const float* __restrict__ edge_sh, const float* __restrict__ edge_fea,
    const float* __restrict__ ele, const int* __restrict__ edge_index,
    const float* __restrict__ fc_b1, const float* __restrict__ fc_b2,
    const float* __restrict__ fc_b3,
    const _Float16* __restrict__ wts,
    float* __restrict__ zsum, float* __restrict__ vsum, int E) {
  __shared__ _Float16 s1h[16 * 192];
  __shared__ _Float16 v1h[3 * 16 * 96];
  __shared__ _Float16 ddh[16 * 96];
  __shared__ _Float16 eleh[16 * 128];
  __shared__ _Float16 hh[16 * 64];
  __shared__ float    wg[16 * 96];
  __shared__ float    shs[16 * 4];
  __shared__ int      idxI[16], idxJ[16];

  const int lane = threadIdx.x;
  const long eb = (long)blockIdx.x * 16;
  int rem = E - (int)eb;
  const int cnt = rem < 16 ? rem : 16;

  if (lane < 16) {
    if (lane < cnt) {
      idxI[lane] = edge_index[eb + lane];
      idxJ[lane] = edge_index[(size_t)E + eb + lane];
    } else { idxI[lane] = 0; idxJ[lane] = 0; }
  }
  for (int t = lane; t < 64; t += 32) {
    int e = t >> 2;
    shs[t] = (e < cnt) ? edge_sh[(eb + e) * 4 + (t & 3)] : 0.f;
  }
  __syncthreads();

  // ---- stage s1 = [s_i, s_j, es] as f16 ----
  for (int t = lane; t < 16 * 192; t += 32) {
    int e = t / 192, k = t - e * 192;
    float v = 0.f;
    if (e < cnt) {
      if (k < 64)       v = s_lin[(size_t)idxI[e] * MS + k];
      else if (k < 128) v = s_lin[(size_t)idxJ[e] * MS + (k - 64)];
      else              v = edge_fea[(eb + e) * DN + (k - 128)];
    }
    s1h[t] = (_Float16)v;
  }
  // ---- stage v1 components + dd = (v1 . sh1)/sqrt(3) ----
  for (int t = lane; t < 16 * 96; t += 32) {
    int e = t / 96, u = t - e * 96;
    float c0 = 0.f, c1 = 0.f, c2 = 0.f;
    if (e < cnt) {
      if (u < 32)      { size_t b = (size_t)idxI[e] * 96 + u * 3;        c0 = v_lin[b]; c1 = v_lin[b + 1]; c2 = v_lin[b + 2]; }
      else if (u < 64) { size_t b = (size_t)idxJ[e] * 96 + (u - 32) * 3; c0 = v_lin[b]; c1 = v_lin[b + 1]; c2 = v_lin[b + 2]; }
      else             { size_t b = (size_t)(eb + e) * DN + 64 + (u - 64) * 3; c0 = edge_fea[b]; c1 = edge_fea[b + 1]; c2 = edge_fea[b + 2]; }
    }
    float dd = (c0 * shs[e * 4 + 1] + c1 * shs[e * 4 + 2] + c2 * shs[e * 4 + 3]) * 0.5773502691896258f;
    ddh[t] = (_Float16)dd;
    v1h[t] = (_Float16)c0;
    v1h[1536 + t] = (_Float16)c1;
    v1h[3072 + t] = (_Float16)c2;
  }
  for (int t = lane; t < 16 * 128; t += 32) {
    int e = t >> 7, k = t & 127;
    eleh[t] = (_Float16)((e < cnt) ? ele[(eb + e) * FCIN + k] : 0.f);
  }
  __syncthreads();

  const int n = lane & 15, rb = (lane >> 4) * 8;

  // ================= FC chain: 128 -> 64 -> 64 -> 96 =================
  {
    v8f h1[4] = {};
    for (int kt = 0; kt < 4; ++kt) {
      v16h a = frag_A(eleh, 128, kt * 32, lane);
#pragma unroll
      for (int nt = 0; nt < 4; ++nt)
        h1[nt] = WMMA_F16(a, frag_B(wts + O_W1, 128, nt * 16, kt * 32, lane), h1[nt]);
    }
#pragma unroll
    for (int nt = 0; nt < 4; ++nt)
#pragma unroll
      for (int m = 0; m < 8; ++m) {
        int c = nt * 16 + n;
        float x = h1[nt][m] + fc_b1[c];
        x = x / (1.f + __expf(-x));
        hh[(rb + m) * 64 + c] = (_Float16)x;
      }
  }
  __syncthreads();
  {
    v8f h2[4] = {};
    for (int kt = 0; kt < 2; ++kt) {
      v16h a = frag_A(hh, 64, kt * 32, lane);
#pragma unroll
      for (int nt = 0; nt < 4; ++nt)
        h2[nt] = WMMA_F16(a, frag_B(wts + O_W2, 64, nt * 16, kt * 32, lane), h2[nt]);
    }
    __syncthreads();
#pragma unroll
    for (int nt = 0; nt < 4; ++nt)
#pragma unroll
      for (int m = 0; m < 8; ++m) {
        int c = nt * 16 + n;
        float x = h2[nt][m] + fc_b2[c];
        x = x / (1.f + __expf(-x));
        hh[(rb + m) * 64 + c] = (_Float16)x;
      }
  }
  __syncthreads();
  {
    v8f wk[6] = {};
    for (int kt = 0; kt < 2; ++kt) {
      v16h a = frag_A(hh, 64, kt * 32, lane);
#pragma unroll
      for (int nt = 0; nt < 6; ++nt)
        wk[nt] = WMMA_F16(a, frag_B(wts + O_W3, 64, nt * 16, kt * 32, lane), wk[nt]);
    }
#pragma unroll
    for (int nt = 0; nt < 6; ++nt)
#pragma unroll
      for (int m = 0; m < 8; ++m) {
        int c = nt * 16 + n;
        wg[(rb + m) * 96 + c] = wk[nt][m] + fc_b3[c];
      }
  }
  __syncthreads();

  // ================= message GEMMs =================
  v8f zsA[4] = {}, zsB[4] = {}, zgA[2] = {}, zgB[2] = {}, svp[2] = {};
  v8f vs0[2] = {}, vs1[2] = {}, vs2[2] = {};

  for (int kt = 0; kt < 6; ++kt) {   // K = 192, shared A fragments (s1)
    v16h a = frag_A(s1h, 192, kt * 32, lane);
#pragma unroll
    for (int nt = 0; nt < 4; ++nt)
      zsA[nt] = WMMA_F16(a, frag_B(wts + O_WSS0, 192, nt * 16, kt * 32, lane), zsA[nt]);
#pragma unroll
    for (int nt = 0; nt < 2; ++nt)
      zgA[nt] = WMMA_F16(a, frag_B(wts + O_WSSG, 192, nt * 16, kt * 32, lane), zgA[nt]);
#pragma unroll
    for (int nt = 0; nt < 2; ++nt)
      svp[nt] = WMMA_F16(a, frag_B(wts + O_WSV1, 192, nt * 16, kt * 32, lane), svp[nt]);
  }
  for (int kt = 0; kt < 3; ++kt) {   // K = 96 (dd and v1 components)
    v16h a = frag_A(ddh, 96, kt * 32, lane);
#pragma unroll
    for (int nt = 0; nt < 4; ++nt)
      zsB[nt] = WMMA_F16(a, frag_B(wts + O_WVV0, 96, nt * 16, kt * 32, lane), zsB[nt]);
#pragma unroll
    for (int nt = 0; nt < 2; ++nt)
      zgB[nt] = WMMA_F16(a, frag_B(wts + O_WVVG, 96, nt * 16, kt * 32, lane), zgB[nt]);
    v16h a0 = frag_A(v1h, 96, kt * 32, lane);
    v16h a1 = frag_A(v1h + 1536, 96, kt * 32, lane);
    v16h a2 = frag_A(v1h + 3072, 96, kt * 32, lane);
#pragma unroll
    for (int nt = 0; nt < 2; ++nt) {
      vs0[nt] = WMMA_F16(a0, frag_B(wts + O_WVS1, 96, nt * 16, kt * 32, lane), vs0[nt]);
      vs1[nt] = WMMA_F16(a1, frag_B(wts + O_WVS1, 96, nt * 16, kt * 32, lane), vs1[nt]);
      vs2[nt] = WMMA_F16(a2, frag_B(wts + O_WVS1, 96, nt * 16, kt * 32, lane), vs2[nt]);
    }
  }

  // ================= combine + atomic segment sum =================
  const float invfan = 0.05892556509887896f;   // 1/sqrt(288)
#pragma unroll
  for (int nt = 0; nt < 4; ++nt)
#pragma unroll
    for (int m = 0; m < 8; ++m) {
      int r = rb + m;
      if (r < cnt) {
        int c = nt * 16 + n;
        float z = (shs[r * 4] * zsA[nt][m] + zsB[nt][m]) * invfan;
        z = z / (1.f + __expf(-z));            // silu
        z *= wg[r * 96 + c];
        atomicAdd(&zsum[(size_t)idxI[r] * MS + c], z);
      }
    }
#pragma unroll
  for (int nt = 0; nt < 2; ++nt)
#pragma unroll
    for (int m = 0; m < 8; ++m) {
      int r = rb + m;
      if (r < cnt) {
        int c = nt * 16 + n;
        float s0 = shs[r * 4];
        float g = (s0 * zgA[nt][m] + zgB[nt][m]) * invfan;
        g = 1.f / (1.f + __expf(-g));
        float wv = wg[r * 96 + 64 + c] * g * invfan;
        float sv = svp[nt][m];
        float o0 = (sv * shs[r * 4 + 1] + vs0[nt][m] * s0) * wv;
        float o1 = (sv * shs[r * 4 + 2] + vs1[nt][m] * s0) * wv;
        float o2 = (sv * shs[r * 4 + 3] + vs2[nt][m] * s0) * wv;
        size_t b = (size_t)idxI[r] * 96 + c * 3;
        atomicAdd(&vsum[b], o0);
        atomicAdd(&vsum[b + 1], o1);
        atomicAdd(&vsum[b + 2], o2);
      }
    }
}

// ---------------------------------------------------------------------------
// Node post-stage: pp projection + self-connection + group-stat accumulation
// ---------------------------------------------------------------------------
__global__ __launch_bounds__(96) void node_post(
    const float* __restrict__ zsum, const float* __restrict__ vsum,
    const float* __restrict__ pp_Ws, const float* __restrict__ pp_bs,
    const float* __restrict__ pp_Wv,
    const float* __restrict__ sc_s, const float* __restrict__ sc_v,
    const int* __restrict__ batch,
    float* __restrict__ ns_buf, float* __restrict__ nv_buf,
    float* __restrict__ gsum) {
  int nn = blockIdx.x, t = threadIdx.x;
  __shared__ float zs_[MS];
  __shared__ float vs_[96];
  if (t < MS) zs_[t] = zsum[(size_t)nn * MS + t];
  vs_[t] = vsum[(size_t)nn * 96 + t];
  __syncthreads();
  int g = batch[nn];
  if (t < MS) {
    float a = 0.f;
    for (int u = 0; u < MS; ++u) a += zs_[u] * pp_Ws[u * MS + t];
    a = a * 0.125f + pp_bs[t] + sc_s[(size_t)nn * MS + t];
    ns_buf[(size_t)nn * MS + t] = a;
    atomicAdd(&gsum[g * 4 + 0], a * (1.f / 64.f));
    atomicAdd(&gsum[g * 4 + 1], a * a * (1.f / 64.f));
  }
  {
    int w = t / 3, i = t - 3 * w;
    float b = 0.f;
    for (int u = 0; u < MV; ++u) b += vs_[u * 3 + i] * pp_Wv[u * MV + w];
    b = b * 0.17677669529663689f + sc_v[(size_t)nn * 96 + t];
    nv_buf[(size_t)nn * 96 + t] = b;
    atomicAdd(&gsum[g * 4 + 2], b * b * (1.f / 96.f));
  }
  if (t == 0) atomicAdd(&gsum[g * 4 + 3], 1.f);
}

__global__ void group_stats(const float* __restrict__ gsum, float* __restrict__ gstat) {
  int g = threadIdx.x;
  if (g < NGRP) {
    float c = fmaxf(gsum[g * 4 + 3], 1.f);
    float mu = gsum[g * 4 + 0] / c;
    float vs = gsum[g * 4 + 1] / c - mu * mu;
    float vv = gsum[g * 4 + 2] / c;
    gstat[g * 3 + 0] = mu;
    gstat[g * 3 + 1] = rsqrtf(vs + EPSL);
    gstat[g * 3 + 2] = rsqrtf(vv + EPSL);
  }
}

__global__ __launch_bounds__(160) void finalize(
    const float* __restrict__ node_fea, const int* __restrict__ batch,
    const float* __restrict__ ns_buf, const float* __restrict__ nv_buf,
    const float* __restrict__ gstat,
    const float* __restrict__ ln_w_s, const float* __restrict__ ln_b_s,
    const float* __restrict__ ln_w_v, float* __restrict__ out) {
  int nn = blockIdx.x, t = threadIdx.x;
  int g = batch[nn];
  float mu = gstat[g * 3], is = gstat[g * 3 + 1], iv = gstat[g * 3 + 2];
  size_t b = (size_t)nn * DN;
  if (t < MS) {
    out[b + t] = node_fea[b + t] + (ns_buf[(size_t)nn * MS + t] - mu) * is * ln_w_s[t] + ln_b_s[t];
  } else {
    int t2 = t - MS;
    out[b + t] = node_fea[b + t] + nv_buf[(size_t)nn * 96 + t2] * iv * ln_w_v[t2 / 3];
  }
}

// ---------------------------------------------------------------------------
// Launcher
// ---------------------------------------------------------------------------
extern "C" void kernel_launch(void* const* d_in, const int* in_sizes, int n_in,
                              void* d_out, int out_size, void* d_ws, size_t ws_size,
                              hipStream_t stream) {
  const float* node_fea = (const float*)d_in[0];
  const float* one_hot  = (const float*)d_in[1];
  const float* edge_sh  = (const float*)d_in[2];
  const float* edge_fea = (const float*)d_in[3];
  const float* ele      = (const float*)d_in[4];
  const int*   edge_idx = (const int*)d_in[5];
  const int*   batch    = (const int*)d_in[6];
  const float* lp_Ws = (const float*)d_in[7];
  const float* lp_bs = (const float*)d_in[8];
  const float* lp_Wv = (const float*)d_in[9];
  const float* pp_Ws = (const float*)d_in[10];
  const float* pp_bs = (const float*)d_in[11];
  const float* pp_Wv = (const float*)d_in[12];
  const float* sc_Ws = (const float*)d_in[13];
  const float* sc_Wv = (const float*)d_in[14];
  const float* W_ss0 = (const float*)d_in[15];
  const float* W_vv0 = (const float*)d_in[16];
  const float* W_ssg = (const float*)d_in[17];
  const float* W_vvg = (const float*)d_in[18];
  const float* W_sv1 = (const float*)d_in[19];
  const float* W_vs1 = (const float*)d_in[20];
  const float* fc_W1 = (const float*)d_in[21];
  const float* fc_b1 = (const float*)d_in[22];
  const float* fc_W2 = (const float*)d_in[23];
  const float* fc_b2 = (const float*)d_in[24];
  const float* fc_W3 = (const float*)d_in[25];
  const float* fc_b3 = (const float*)d_in[26];
  const float* ln_w_s = (const float*)d_in[27];
  const float* ln_b_s = (const float*)d_in[28];
  const float* ln_w_v = (const float*)d_in[29];

  const int N = in_sizes[0] / DN;
  const int E = in_sizes[3] / DN;

  // Workspace layout (f32 unless noted)
  float* ws = (float*)d_ws;
  float* s_lin  = ws;                       // N*64
  float* v_lin  = s_lin + (size_t)N * 64;   // N*96
  float* sc_s_w = v_lin + (size_t)N * 96;   // N*64
  float* sc_v_w = sc_s_w + (size_t)N * 64;  // N*96
  float* zsum   = sc_v_w + (size_t)N * 96;  // N*64  (zeroed)
  float* vsum   = zsum + (size_t)N * 64;    // N*96  (zeroed)
  float* gsum   = vsum + (size_t)N * 96;    // 64    (zeroed)
  float* gstat  = gsum + 64;                // 48 (+pad)
  float* ns_buf = gstat + 64;               // N*64
  float* nv_buf = ns_buf + (size_t)N * 64;  // N*96
  _Float16* wts = (_Float16*)(nv_buf + (size_t)N * 96);  // W_TOTAL halves

  // zero accumulators (zsum, vsum, gsum contiguous)
  hipMemsetAsync(zsum, 0, ((size_t)N * 160 + 64) * sizeof(float), stream);

  // weight transpose -> f16  [Nout][K]
  {
    struct { const float* s; int off; int K; int Nn; } T[9] = {
      {W_ss0, O_WSS0, 192, 64}, {W_vv0, O_WVV0, 96, 64},
      {W_ssg, O_WSSG, 192, 32}, {W_vvg, O_WVVG, 96, 32},
      {W_sv1, O_WSV1, 192, 32}, {W_vs1, O_WVS1, 96, 32},
      {fc_W1, O_W1, 128, 64},   {fc_W2, O_W2, 64, 64},
      {fc_W3, O_W3, 64, 96},
    };
    for (int i = 0; i < 9; ++i) {
      int tot = T[i].K * T[i].Nn;
      w2h_transpose<<<(tot + 255) / 256, 256, 0, stream>>>(T[i].s, wts + T[i].off, T[i].K, T[i].Nn);
    }
  }

  node_pre<<<N, 96, 0, stream>>>(node_fea, one_hot, lp_Ws, lp_bs, lp_Wv, sc_Ws, sc_Wv,
                                 s_lin, v_lin, sc_s_w, sc_v_w);

  edge_kernel<<<(E + 15) / 16, 32, 0, stream>>>(s_lin, v_lin, edge_sh, edge_fea, ele,
                                                edge_idx, fc_b1, fc_b2, fc_b3, wts,
                                                zsum, vsum, E);

  node_post<<<N, 96, 0, stream>>>(zsum, vsum, pp_Ws, pp_bs, pp_Wv, sc_s_w, sc_v_w,
                                  batch, ns_buf, nv_buf, gsum);

  group_stats<<<1, 32, 0, stream>>>(gsum, gstat);

  finalize<<<N, 160, 0, stream>>>(node_fea, batch, ns_buf, nv_buf, gstat,
                                  ln_w_s, ln_b_s, ln_w_v, (float*)d_out);
}
